// QuantizedLinearZSE_30855045054748
// MI455X (gfx1250) — matrix operand order repaired and checked by reading, not compiled, not run on hardware
//
#include <hip/hip_runtime.h>

typedef __attribute__((ext_vector_type(16))) _Float16     v16h;
typedef __attribute__((ext_vector_type(2)))  _Float16     h2;
typedef __attribute__((ext_vector_type(8)))  float        v8f;
typedef __attribute__((ext_vector_type(2)))  float        f32x2;
typedef __attribute__((ext_vector_type(4)))  float        f32x4;
typedef __attribute__((ext_vector_type(4)))  int          i32x4;
typedef __attribute__((ext_vector_type(4)))  unsigned int u32x4;
typedef __attribute__((ext_vector_type(8)))  unsigned int u32x8;

#define K_DIM   4096
#define N_DIM   4096
#define GROUPS  32          // 4096 / 128
#define PKROW   2048        // packed int32 (one byte each) per output row
#define BK      64          // K per staging step (2 WMMA K-steps)
#define KTILES  (K_DIM / BK)   // 64
#define LDSW    72          // uint16 per LDS row: 64 data + 8 pad (144B, 16B-aligned)

// two f32 -> packed f16 pair (v_cvt_pk_f16_f32)
static __device__ __forceinline__ unsigned int pkh2(float lo, float hi) {
    f32x2 v = {lo, hi};
    h2 h = __builtin_convertvector(v, h2);
    return __builtin_bit_cast(unsigned int, h);
}

// int4 pair -> packed f16 pair via mantissa-magic:
// 0x6400 | n bitcasts to f16 (1024+n); subtract 1032 -> (n-8); then * scale.
static __device__ __forceinline__ unsigned int dq2(int p, h2 scpair) {
    unsigned int u = 0x64006400u | (unsigned)(p & 15) | ((unsigned)(p & 0xF0) << 12);
    h2 h = __builtin_bit_cast(h2, u);
    h2 magic = {(_Float16)1032.0f, (_Float16)1032.0f};
    h2 r = (h - magic) * scpair;            // v_pk_add_f16 + v_pk_mul_f16
    return __builtin_bit_cast(unsigned int, r);
}

// A fragment (16x32 f16, ISA 7.12.2): lanes0-15 dwords hold K0..7 then K16..23,
// lanes16-31 hold K8..15 then K24..31. `row` points at the 32-f16 K-block.
static __device__ __forceinline__ v16h frag_a(const unsigned short* row, int hsel) {
    u32x4 q0 = *(const u32x4*)(row + hsel * 8);
    u32x4 q1 = *(const u32x4*)(row + hsel * 8 + 16);
    u32x8 t;
    t[0] = q0[0]; t[1] = q0[1]; t[2] = q0[2]; t[3] = q0[3];
    t[4] = q1[0]; t[5] = q1[1]; t[6] = q1[2]; t[7] = q1[3];
    return __builtin_bit_cast(v16h, t);
}
// B fragment (32x16 f16, same K order as A): lanes0-15 hold K0..7 then K16..23,
// lanes16-31 hold K8..15 then K24..31.
static __device__ __forceinline__ v16h frag_b(const unsigned short* row, int hsel) {
    u32x4 q0 = *(const u32x4*)(row + hsel * 8);
    u32x4 q1 = *(const u32x4*)(row + hsel * 8 + 16);
    u32x8 t;
    t[0] = q0[0]; t[1] = q0[1]; t[2] = q0[2]; t[3] = q0[3];
    t[4] = q1[0]; t[5] = q1[1]; t[6] = q1[2]; t[7] = q1[3];
    return __builtin_bit_cast(v16h, t);
}

__global__ __launch_bounds__(256)
void QuantizedLinearZSE_30855045054748_kernel(const float* __restrict__ x,
                                              const int*   __restrict__ wp,
                                              const float* __restrict__ scales,
                                              float*       __restrict__ out)
{
    __shared__ __align__(16) unsigned short sA[128 * LDSW];  // x tile, f16
    __shared__ __align__(16) unsigned short sB[128 * LDSW];  // dequantized W tile, f16

    const int tid  = threadIdx.x;
    const int lane = tid & 31;
    const int wave = tid >> 5;
    const int l16  = lane & 15;
    const int hsel = lane >> 4;

    const int mBlk = blockIdx.y;   // token-tile
    const int nBlk = blockIdx.x;   // out-feature tile

    const int wmBase = (wave & 1) * 64;   // wave's M offset in tile
    const int wnBase = (wave >> 1) * 32;  // wave's N offset in tile

    // ---- cooperative staging: 2 threads per row, each does half the 64-wide K step ----
    const int ra   = tid >> 1;   // 0..127 row within tile
    const int half = tid & 1;    // which 32-wide K half

    const f32x4* xv = (const f32x4*)x +
                      (size_t)(mBlk * 128 + ra) * (K_DIM / 4) + half * 8;
    const i32x4* wv = (const i32x4*)wp +
                      (size_t)(nBlk * 128 + ra) * (PKROW / 4) + half * 4;
    const float* sp = scales + (size_t)(nBlk * 128 + ra) * GROUPS;

    // prologue: load K-step 0 into registers
    f32x4 xr[8];
    i32x4 wr[4];
#pragma unroll
    for (int v = 0; v < 8; ++v) xr[v] = xv[v];
#pragma unroll
    for (int v = 0; v < 4; ++v) wr[v] = wv[v];
    float sc = sp[0];            // group 0 scale; reloaded at group boundaries

    v8f acc[4][2];
#pragma unroll
    for (int i = 0; i < 4; ++i)
#pragma unroll
        for (int j = 0; j < 2; ++j)
            acc[i][j] = (v8f){0.f, 0.f, 0.f, 0.f, 0.f, 0.f, 0.f, 0.f};

    for (int kt = 0; kt < KTILES; ++kt) {
        __syncthreads();   // previous step's LDS reads done

        // ---- stage A: f32 -> f16, 64 contiguous bytes ----
        unsigned short* aRow = &sA[ra * LDSW + half * 32];
#pragma unroll
        for (int q = 0; q < 4; ++q) {
            u32x4 apk;
            apk[0] = pkh2(xr[2 * q][0], xr[2 * q][1]);
            apk[1] = pkh2(xr[2 * q][2], xr[2 * q][3]);
            apk[2] = pkh2(xr[2 * q + 1][0], xr[2 * q + 1][1]);
            apk[3] = pkh2(xr[2 * q + 1][2], xr[2 * q + 1][3]);
            *(u32x4*)(aRow + 8 * q) = apk;
        }

        // ---- stage B: int4 magic dequant -> f16, 64 contiguous bytes ----
        {
            _Float16 hsc = (_Float16)sc;
            h2 scpair = {hsc, hsc};
            unsigned short* bRow = &sB[ra * LDSW + half * 32];
#pragma unroll
            for (int q = 0; q < 4; ++q) {
                u32x4 bpk;
                bpk[0] = dq2(wr[q][0], scpair);
                bpk[1] = dq2(wr[q][1], scpair);
                bpk[2] = dq2(wr[q][2], scpair);
                bpk[3] = dq2(wr[q][3], scpair);
                *(u32x4*)(bRow + 8 * q) = bpk;
            }
        }

        __syncthreads();   // step fully staged

        // ---- software pipeline: issue next step's global loads now; they stay
        //      outstanding under the 16 WMMAs below ----
        if (kt + 1 < KTILES) {
            const f32x4* xn = xv + (size_t)(kt + 1) * 16;
#pragma unroll
            for (int v = 0; v < 8; ++v) xr[v] = xn[v];
            const i32x4* wn = wv + (size_t)(kt + 1) * 8;
#pragma unroll
            for (int v = 0; v < 4; ++v) wr[v] = wn[v];
            if (((kt + 1) & 1) == 0)            // uniform: new 128-wide K group
                sc = sp[(kt + 1) >> 1];
        }

        // ---- fragments + WMMA: two 32-wide K-steps ----
#pragma unroll
        for (int ks = 0; ks < 2; ++ks) {
            v16h af[4], bfr[2];
#pragma unroll
            for (int i = 0; i < 4; ++i)
                af[i] = frag_a(&sA[(wmBase + i * 16 + l16) * LDSW + ks * 32], hsel);
#pragma unroll
            for (int j = 0; j < 2; ++j)
                bfr[j] = frag_b(&sB[(wnBase + j * 16 + l16) * LDSW + ks * 32], hsel);

#pragma unroll
            for (int i = 0; i < 4; ++i)
#pragma unroll
                for (int j = 0; j < 2; ++j)
                    acc[i][j] = __builtin_amdgcn_wmma_f32_16x16x32_f16(
                        /*neg_a=*/false, af[i], /*neg_b=*/false, bfr[j],
                        /*c_mod=*/(short)0, acc[i][j],
                        /*reuse_a=*/false, /*reuse_b=*/false);
        }
    }

    // ---- store C: VGPR r holds M=r (lanes0-15) / M=8+r (lanes16-31), N=l16 ----
    const int colBase = nBlk * 128 + wnBase + l16;
#pragma unroll
    for (int i = 0; i < 4; ++i) {
        const int rowBase = mBlk * 128 + wmBase + i * 16 + hsel * 8;
#pragma unroll
        for (int j = 0; j < 2; ++j) {
            const int c = colBase + j * 16;
#pragma unroll
            for (int r = 0; r < 8; ++r)
                out[(size_t)(rowBase + r) * N_DIM + c] = acc[i][j][r];
        }
    }
}

extern "C" void kernel_launch(void* const* d_in, const int* in_sizes, int n_in,
                              void* d_out, int out_size, void* d_ws, size_t ws_size,
                              hipStream_t stream) {
    (void)n_in; (void)out_size; (void)d_ws; (void)ws_size;
    const float* x      = (const float*)d_in[0];
    const int*   wp     = (const int*)d_in[1];
    const float* scales = (const float*)d_in[2];
    // d_in[3] = group_size scalar (128, hardcoded)
    float* out = (float*)d_out;

    const int M = in_sizes[0] / K_DIM;          // tokens (8192)
    dim3 grid(N_DIM / 128, M / 128, 1);
    QuantizedLinearZSE_30855045054748_kernel<<<grid, 256, 0, stream>>>(x, wp, scales, out);
}
